// PSGAN_Generator_39539468927734
// MI455X (gfx1250) — compile-verified, matrix operand order checked by reading, not run on hardware
//
#include <hip/hip_runtime.h>
#include <hip/hip_bf16.h>
#include <math.h>

// ---------------------------------------------------------------------------
// PSGAN generator forward on CDNA5 (gfx1250, wave32).
// Conv / transposed-conv as implicit-im2col WMMA GEMMs, 64x64 block tiles,
// 4 accumulators per wave (A-fragment reuse), async global->LDS weight staging,
// loop-invariant im2col predicates (32-bit index math only).
// ---------------------------------------------------------------------------

#define BATCH 64
#define TWO_PI_F 6.283185307179586f

typedef _Float16 v16h __attribute__((ext_vector_type(16)));
typedef _Float16 h8   __attribute__((ext_vector_type(8)));
typedef _Float16 h4   __attribute__((ext_vector_type(4)));
typedef float    v8f  __attribute__((ext_vector_type(8)));

union AFrag { v16h v; h8 h[2]; };

// --- CDNA5 async global->LDS path (signature probed in earlier rounds) ------
#if defined(__has_builtin)
#  if __has_builtin(__builtin_amdgcn_global_load_async_to_lds_b128) && \
      __has_builtin(__builtin_amdgcn_s_wait_asynccnt)
#    define USE_ASYNC_LDS 1
#  endif
#endif

#ifdef USE_ASYNC_LDS
typedef int v4i_vs __attribute__((__vector_size__(16)));
typedef __attribute__((address_space(1))) v4i_vs* g_v4i;
typedef __attribute__((address_space(3))) v4i_vs* l_v4i;
#  define ASYNC_COPY16(gsrc, ldst)                                             \
      __builtin_amdgcn_global_load_async_to_lds_b128(                          \
          (g_v4i)(size_t)(gsrc), (l_v4i)(ldst), 0, 0)
#  define ASYNC_WAIT() __builtin_amdgcn_s_wait_asynccnt(0)
#else
#  define ASYNC_COPY16(gsrc, ldst) (*(h8*)(ldst) = *(const h8*)(gsrc))
#  define ASYNC_WAIT() ((void)0)
#endif

// ---------------------------------------------------------------------------
// elementwise helpers
// ---------------------------------------------------------------------------
__global__ void cvt_f32_to_f16(const float* __restrict__ x, _Float16* __restrict__ y, int n) {
    int i = blockIdx.x * blockDim.x + threadIdx.x;
    if (i < n) y[i] = (_Float16)x[i];
}

__global__ void tanh_to_f16(const float* __restrict__ x, _Float16* __restrict__ y, int n) {
    int i = blockIdx.x * blockDim.x + threadIdx.x;
    if (i < n) y[i] = (_Float16)tanhf(x[i]);
}

// ---------------------------------------------------------------------------
// Per-channel batch-norm statistics over (B, H, W).  One block per channel.
// ---------------------------------------------------------------------------
__global__ __launch_bounds__(256)
void bn_stats(const float* __restrict__ x, float* __restrict__ stats, int C, int HW, int lgHW) {
    int c = blockIdx.x;
    int N = BATCH * HW;
    float s = 0.f, sq = 0.f;
    for (int t = threadIdx.x; t < N; t += 256) {
        int b = t >> lgHW;
        int p = t & (HW - 1);
        float v = x[((size_t)b * C + c) * HW + p];
        s += v; sq += v * v;
    }
    __shared__ float rs[256], rq[256];
    rs[threadIdx.x] = s; rq[threadIdx.x] = sq;
    __syncthreads();
    for (int o = 128; o > 0; o >>= 1) {
        if (threadIdx.x < (unsigned)o) {
            rs[threadIdx.x] += rs[threadIdx.x + o];
            rq[threadIdx.x] += rq[threadIdx.x + o];
        }
        __syncthreads();
    }
    if (threadIdx.x == 0) {
        float inv = 1.f / (float)N;
        float m = rs[0] * inv;
        float var = rq[0] * inv - m * m;
        stats[c * 2 + 0] = m;
        stats[c * 2 + 1] = rsqrtf(var + 1e-5f);
    }
}

// ---------------------------------------------------------------------------
// Fused BN + activation + cast-to-f16 (C is a power of two here: 64/128/256)
// ---------------------------------------------------------------------------
__global__ void bn_act_f16(const float* __restrict__ x, const float* __restrict__ stats,
                           const float* __restrict__ g, const float* __restrict__ be,
                           _Float16* __restrict__ y, int C, int lgHW, int leaky, int total) {
    int i = blockIdx.x * blockDim.x + threadIdx.x;
    if (i >= total) return;
    int c = (i >> lgHW) & (C - 1);
    float m = stats[c * 2 + 0], istd = stats[c * 2 + 1];
    float v = (x[i] - m) * istd * g[c] + be[c];
    v = v > 0.f ? v : (leaky ? 0.2f * v : 0.f);
    y[i] = (_Float16)v;
}

// ---------------------------------------------------------------------------
// Strided conv (stride 2, k=4, pad 1, square) as WMMA GEMM.
//   M = B*Hout^2 (row -> b,oy,ox),  N = Cout,  K = Cin*16, k = ic*16 + kh*4+kw
// Weight f16 layout (Cout,Cin,4,4): B[k][oc] = W16[oc*Kreal + k]  (k-contiguous)
// Block: 128 thr = 4 waves; tile 64(M) x 64(N); 4 accumulators per wave.
// Each thread's A slot (16 halves) is exactly one input channel's 4x4 window:
// row-level iy predicate, edge predicates only for kw=0/3 (loop-invariant).
// ---------------------------------------------------------------------------
__global__ __launch_bounds__(128)
void conv_gemm_wmma(const _Float16* __restrict__ X, const _Float16* __restrict__ W16,
                    const float* __restrict__ bias, float* __restrict__ Y,
                    int Cin, int Cout, int Hin, int lgHo) {
    __shared__ _Float16 At[64][32];   // [m_local][k]
    __shared__ _Float16 Bt[64][32];   // [n][k]  (column-major B)

    const int tid  = threadIdx.x;
    const int lane = tid & 31;
    const int wv   = tid >> 5;

    const int Hout   = Hin >> 1;
    const int homask = Hout - 1;
    const int HinHin = Hin * Hin;
    const int m0     = blockIdx.x * 64;
    const int oc0    = blockIdx.y * 64;
    const int Kreal  = Cin * 16;
    const int nK     = (Kreal + 31) >> 5;

    // --- A staging: fixed slot, loop-invariant coordinates/predicates -------
    const int arow = tid >> 1;            // A row this thread stages (0..63)
    const int aj0  = (tid & 1) * 16;      // 16-half group => exactly one ic
    const int am   = m0 + arow;
    const int aox  = am & homask;
    const int at_  = am >> lgHo;
    const int aoy  = at_ & homask;
    const int ab   = at_ >> lgHo;
    const _Float16* xp = X + (size_t)ab * Cin * HinHin;   // batch base
    const int aiy0 = (aoy << 1) - 1;
    const int aix0 = (aox << 1) - 1;
    const bool ixok0 = (unsigned)aix0       < (unsigned)Hin;  // kw=0 edge
    const bool ixok3 = (unsigned)(aix0 + 3) < (unsigned)Hin;  // kw=3 edge

    // --- B staging slot ------------------------------------------------------
    const int brow = tid >> 1;
    const int bj0  = (tid & 1) * 16;
    const int boc  = oc0 + brow;
    const _Float16* wrow = W16 + (size_t)boc * Kreal;

    v8f acc[4] = {};

    for (int kc = 0; kc < nK; ++kc) {
        const int kbase = kc << 5;

        // ---- stage A: one channel's 4x4 window (4 x h4 vector stores) ----
        {
            const int kg = kbase + aj0;
            if (kg < Kreal) {
                const _Float16* cp = xp + (kg >> 4) * HinHin + aix0;
                #pragma unroll
                for (int kh = 0; kh < 4; ++kh) {
                    int iy = aiy0 + kh;
                    h4 r = {};
                    if ((unsigned)iy < (unsigned)Hin) {
                        const _Float16* rp = cp + iy * Hin;
                        if (ixok0) r.x = rp[0];
                        r.y = rp[1];
                        r.z = rp[2];
                        if (ixok3) r.w = rp[3];
                    }
                    *(h4*)&At[arow][aj0 + (kh << 2)] = r;
                }
            } else {
                h8 z = {};
                *(h8*)&At[arow][aj0]     = z;
                *(h8*)&At[arow][aj0 + 8] = z;
            }
        }
        // ---- stage B: two 16-byte runs, async global->LDS when legal ----
        #pragma unroll
        for (int s = 0; s < 16; s += 8) {
            int k = kbase + bj0 + s;
            if (boc < Cout && (k + 8) <= Kreal) {
                ASYNC_COPY16(wrow + k, &Bt[brow][bj0 + s]);
            } else {
                h8 z = {};
                if (boc < Cout) {
                    #pragma unroll
                    for (int j = 0; j < 8; ++j)
                        if (k + j < Kreal) z[j] = wrow[k + j];
                }
                *(h8*)&Bt[brow][bj0 + s] = z;
            }
        }
        ASYNC_WAIT();
        __syncthreads();

        // ---- fragments + 4 WMMAs (A reused across the 64-wide N tile) ----
        const int k0 = (lane >> 4) << 3;           // 0 or 8
        AFrag a;
        {
            int r = (wv << 4) + (lane & 15);
            a.h[0] = *reinterpret_cast<const h8*>(&At[r][k0]);
            a.h[1] = *reinterpret_cast<const h8*>(&At[r][k0 + 16]);
        }
        AFrag bf[4];
        #pragma unroll
        for (int j = 0; j < 4; ++j) {
            int nn = (j << 4) + (lane & 15);
            bf[j].h[0] = *reinterpret_cast<const h8*>(&Bt[nn][k0]);
            bf[j].h[1] = *reinterpret_cast<const h8*>(&Bt[nn][k0 + 16]);
        }
        #pragma unroll
        for (int j = 0; j < 4; ++j)
            acc[j] = __builtin_amdgcn_wmma_f32_16x16x32_f16(false, a.v, false, bf[j].v,
                                                            (short)0, acc[j], false, false);
        __syncthreads();
    }

    // ---- epilogue: bias add, scatter to NCHW f32 ----
    const int nn = lane & 15;
    #pragma unroll
    for (int j = 0; j < 4; ++j) {
        int oc = oc0 + (j << 4) + nn;
        if (oc >= Cout) continue;
        float bv = bias[oc];
        #pragma unroll
        for (int r = 0; r < 8; ++r) {
            int ml = (wv << 4) + r + ((lane >> 4) << 3);
            int m  = m0 + ml;
            int ox = m & homask;
            int t  = m >> lgHo;
            int oy = t & homask;
            int b  = t >> lgHo;
            Y[(((size_t)b * Cout + oc) * Hout + oy) * Hout + ox] = acc[j][r] + bv;
        }
    }
}

// ---------------------------------------------------------------------------
// Transposed conv (stride 2, k=4, pad 1, square) as parity-class WMMA GEMM.
// grid.z = (py,px).  Tap (ty,tx) reads input (y+ty-1+py, x+tx-1+px), weight
// w[ic][oc][3-2ty-py][3-2tx-px].  Mc = B*Hin^2, K = Cin*4 (always a multiple
// of 32 here -> no K tail).  Tap offsets/predicates are loop-invariant.
// mode==1: fused tanh epilogue (writes d_out).
// ---------------------------------------------------------------------------
__global__ __launch_bounds__(128)
void deconv_gemm_wmma(const _Float16* __restrict__ X, const _Float16* __restrict__ W16,
                      const float* __restrict__ bias, float* __restrict__ Y,
                      int Cin, int Cout, int Hin, int lgHi, int mode) {
    __shared__ _Float16 At[64][32];
    __shared__ _Float16 Bt[64][32];

    const int tid  = threadIdx.x;
    const int lane = tid & 31;
    const int wv   = tid >> 5;

    const int py = blockIdx.z >> 1;
    const int px = blockIdx.z & 1;
    const int m0  = blockIdx.x * 64;
    const int oc0 = blockIdx.y * 64;
    const int Hout   = Hin << 1;
    const int HinHin = Hin * Hin;
    const int Kreal  = Cin * 4;          // multiple of 32 for all layers here
    const int nK     = Kreal >> 5;
    const int hmask  = Hin - 1;

    // --- A staging: loop-invariant tap offsets + validity ---
    const int arow = tid >> 1;
    const int aj0  = (tid & 1) * 16;
    const int am   = m0 + arow;
    const int ax   = am & hmask;
    const int at_  = am >> lgHi;
    const int ay   = at_ & hmask;
    const int ab   = at_ >> lgHi;
    const _Float16* xp = X + (size_t)ab * Cin * HinHin;
    const int aiy0 = ay - 1 + py;
    const int aix0 = ax - 1 + px;
    int  toff[4];
    bool tok[4];
    #pragma unroll
    for (int t = 0; t < 4; ++t) {
        int iy = aiy0 + (t >> 1);
        int ix = aix0 + (t & 1);
        tok[t]  = ((unsigned)iy < (unsigned)Hin) && ((unsigned)ix < (unsigned)Hin);
        toff[t] = iy * Hin + ix;
    }

    // --- B staging: loop-invariant weight sub-offsets per tap ---
    const int brow = tid >> 1;
    const int bj0  = (tid & 1) * 16;
    const int boc  = oc0 + brow;
    int woff[4];
    #pragma unroll
    for (int t = 0; t < 4; ++t)
        woff[t] = (3 - 2 * (t >> 1) - py) * 4 + (3 - 2 * (t & 1) - px);

    v8f acc[4] = {};

    for (int kc = 0; kc < nK; ++kc) {
        const int kbase = kc << 5;

        // prefetch next weight chunk (global_prefetch_b8)
        if (kc + 1 < nK && boc < Cout) {
            int icn = (kbase + 32) >> 2;
            __builtin_prefetch(&W16[((size_t)icn * Cout + boc) * 16], 0, 0);
        }

        // ---- stage A: 4 channels x 4 taps, predicated scalar loads ----
        {
            const int ic0 = (kbase + aj0) >> 2;
            #pragma unroll
            for (int q = 0; q < 4; ++q) {
                const _Float16* cp = xp + (ic0 + q) * HinHin;
                h4 r = {};
                if (tok[0]) r.x = cp[toff[0]];
                if (tok[1]) r.y = cp[toff[1]];
                if (tok[2]) r.z = cp[toff[2]];
                if (tok[3]) r.w = cp[toff[3]];
                *(h4*)&At[arow][aj0 + (q << 2)] = r;
            }
        }
        // ---- stage B: 4 channels x 4 weight taps ----
        if (boc < Cout) {
            const int ic0 = (kbase + bj0) >> 2;
            #pragma unroll
            for (int q = 0; q < 4; ++q) {
                const _Float16* wp = W16 + ((size_t)(ic0 + q) * Cout + boc) * 16;
                h4 r = { wp[woff[0]], wp[woff[1]], wp[woff[2]], wp[woff[3]] };
                *(h4*)&Bt[brow][bj0 + (q << 2)] = r;
            }
        } else {
            h8 z = {};
            *(h8*)&Bt[brow][bj0]     = z;
            *(h8*)&Bt[brow][bj0 + 8] = z;
        }
        __syncthreads();

        const int k0 = (lane >> 4) << 3;
        AFrag a;
        {
            int r = (wv << 4) + (lane & 15);
            a.h[0] = *reinterpret_cast<const h8*>(&At[r][k0]);
            a.h[1] = *reinterpret_cast<const h8*>(&At[r][k0 + 16]);
        }
        AFrag bf[4];
        #pragma unroll
        for (int j = 0; j < 4; ++j) {
            int nn = (j << 4) + (lane & 15);
            bf[j].h[0] = *reinterpret_cast<const h8*>(&Bt[nn][k0]);
            bf[j].h[1] = *reinterpret_cast<const h8*>(&Bt[nn][k0 + 16]);
        }
        #pragma unroll
        for (int j = 0; j < 4; ++j)
            acc[j] = __builtin_amdgcn_wmma_f32_16x16x32_f16(false, a.v, false, bf[j].v,
                                                            (short)0, acc[j], false, false);
        __syncthreads();
    }

    const int nn = lane & 15;
    #pragma unroll
    for (int j = 0; j < 4; ++j) {
        int oc = oc0 + (j << 4) + nn;
        if (oc >= Cout) continue;
        float bv = bias[oc];
        #pragma unroll
        for (int r = 0; r < 8; ++r) {
            int ml = (wv << 4) + r + ((lane >> 4) << 3);
            int m  = m0 + ml;
            int x  = m & hmask;
            int t  = m >> lgHi;
            int y  = t & hmask;
            int b  = t >> lgHi;
            float val = acc[j][r] + bv;
            if (mode == 1) val = tanhf(val);
            int oy = (y << 1) + py;
            int ox = (x << 1) + px;
            Y[(((size_t)b * Cout + oc) * Hout + oy) * Hout + ox] = val;
        }
    }
}

// ---------------------------------------------------------------------------
// Tiny MLP head: x = relu(zg @ lW.T + lb); K1 = x@l1W.T+l1b; K2 = x@l2W.T+l2b
// ---------------------------------------------------------------------------
__global__ __launch_bounds__(64)
void mlp_wave_params(const float* __restrict__ Zg, const float* __restrict__ lW,
                     const float* __restrict__ lb, const float* __restrict__ l1W,
                     const float* __restrict__ l1b, const float* __restrict__ l2W,
                     const float* __restrict__ l2b, float* __restrict__ K1,
                     float* __restrict__ K2) {
    int b = threadIdx.x;
    float zg[64];
    for (int c = 0; c < 64; ++c)
        zg[c] = Zg[((size_t)b * 64 + c) * 256];   // [b][c][0][0]
    float x[64];
    for (int j = 0; j < 64; ++j) {
        float s = lb[j];
        for (int c = 0; c < 64; ++c) s += zg[c] * lW[j * 64 + c];
        x[j] = s > 0.f ? s : 0.f;
    }
    for (int p = 0; p < 8; ++p) {
        float s1 = l1b[p], s2 = l2b[p];
        for (int j = 0; j < 64; ++j) {
            s1 += x[j] * l1W[p * 64 + j];
            s2 += x[j] * l2W[p * 64 + j];
        }
        K1[b * 8 + p] = s1;
        K2[b * 8 + p] = s2;
    }
}

// ---------------------------------------------------------------------------
// Assemble Z = concat(Z_c, Z_l, Z_g, Z_p) in f16, NCHW (B,168,16,16).
// ---------------------------------------------------------------------------
__global__ void pack_latent(const _Float16* __restrict__ Zc16, const float* __restrict__ Zl,
                            const float* __restrict__ Zg, const float* __restrict__ K1,
                            const float* __restrict__ K2, const float* __restrict__ phi,
                            _Float16* __restrict__ Z16, int total) {
    int i = blockIdx.x * blockDim.x + threadIdx.x;
    if (i >= total) return;
    int s = i & 255;
    int c = (i >> 8) % 168;
    int b = i / (168 * 256);
    _Float16 v;
    if (c < 64) {
        v = Zc16[((size_t)b * 64 + c) * 256 + s];
    } else if (c < 96) {
        v = (_Float16)Zl[((size_t)b * 32 + (c - 64)) * 256 + s];
    } else if (c < 160) {
        v = (_Float16)Zg[((size_t)b * 64 + (c - 96)) * 256 + s];
    } else {
        int p = c - 160;
        int row = s >> 4, col = s & 15;
        float w = K1[b * 8 + p] * (float)row + K2[b * 8 + p] * (float)col
                + phi[b * 8 + p] * TWO_PI_F;
        v = (_Float16)sinf(w);
    }
    Z16[i] = v;
}

// ---------------------------------------------------------------------------
// host orchestration
// ---------------------------------------------------------------------------
static inline int ilog2i(int x) { int l = 0; while ((1 << l) < x) ++l; return l; }

extern "C" void kernel_launch(void* const* d_in, const int* in_sizes, int n_in,
                              void* d_out, int out_size, void* d_ws, size_t ws_size,
                              hipStream_t stream) {
    const float* Zl     = (const float*)d_in[0];
    const float* Zg     = (const float*)d_in[1];
    const float* imgs   = (const float*)d_in[2];
    const float* phi    = (const float*)d_in[3];
    const float* lW     = (const float*)d_in[4];
    const float* lb     = (const float*)d_in[5];
    const float* l1W    = (const float*)d_in[6];
    const float* l1b    = (const float*)d_in[7];
    const float* l2W    = (const float*)d_in[8];
    const float* l2b    = (const float*)d_in[9];
    const float* cod_w0 = (const float*)d_in[10];
    const float* cod_b0 = (const float*)d_in[11];
    const float* cod_g0 = (const float*)d_in[12];
    const float* cod_be0= (const float*)d_in[13];
    const float* cod_w1 = (const float*)d_in[14];
    const float* cod_b1 = (const float*)d_in[15];
    const float* cod_g1 = (const float*)d_in[16];
    const float* cod_be1= (const float*)d_in[17];
    const float* cod_w2 = (const float*)d_in[18];
    const float* cod_b2 = (const float*)d_in[19];
    const float* gen_w0 = (const float*)d_in[20];
    const float* gen_b0 = (const float*)d_in[21];
    const float* gen_g0 = (const float*)d_in[22];
    const float* gen_be0= (const float*)d_in[23];
    const float* gen_w1 = (const float*)d_in[24];
    const float* gen_b1 = (const float*)d_in[25];
    const float* gen_g1 = (const float*)d_in[26];
    const float* gen_be1= (const float*)d_in[27];
    const float* gen_w2 = (const float*)d_in[28];
    const float* gen_b2 = (const float*)d_in[29];

    // --- workspace arena ---------------------------------------------------
    char* base = (char*)d_ws;
    size_t off = 0;
    auto alloc = [&](size_t bytes) -> void* {
        void* p = base + off;
        off = (off + bytes + 255) & ~(size_t)255;
        return p;
    };

    const int n_imgs = BATCH * 3 * 128 * 128;
    _Float16* imgs16 = (_Float16*)alloc((size_t)n_imgs * 2);
    _Float16* cw0    = (_Float16*)alloc((size_t)64 * 48 * 2);
    _Float16* cw1    = (_Float16*)alloc((size_t)128 * 1024 * 2);
    _Float16* cw2    = (_Float16*)alloc((size_t)64 * 2048 * 2);
    _Float16* gw0    = (_Float16*)alloc((size_t)168 * 256 * 16 * 2);
    _Float16* gw1    = (_Float16*)alloc((size_t)256 * 128 * 16 * 2);
    _Float16* gw2    = (_Float16*)alloc((size_t)128 * 3 * 16 * 2);
    float*    raw    = (float*)alloc((size_t)BATCH * 128 * 64 * 64 * 4);   // 134 MB
    _Float16* actA   = (_Float16*)alloc((size_t)BATCH * 128 * 64 * 64 * 2);
    _Float16* actB   = (_Float16*)alloc((size_t)BATCH * 64 * 64 * 64 * 2);
    _Float16* Zc16   = (_Float16*)alloc((size_t)BATCH * 64 * 256 * 2);
    _Float16* Z16    = (_Float16*)alloc((size_t)BATCH * 168 * 256 * 2);
    float*    K1     = (float*)alloc((size_t)BATCH * 8 * 4);
    float*    K2     = (float*)alloc((size_t)BATCH * 8 * 4);
    float*    stats  = (float*)alloc((size_t)256 * 2 * 4);
    (void)ws_size; (void)n_in; (void)in_sizes; (void)out_size;

    auto cvt = [&](const float* s, _Float16* d, int n) {
        cvt_f32_to_f16<<<(n + 255) / 256, 256, 0, stream>>>(s, d, n);
    };

    cvt(imgs, imgs16, n_imgs);
    cvt(cod_w0, cw0, 64 * 48);
    cvt(cod_w1, cw1, 128 * 1024);
    cvt(cod_w2, cw2, 64 * 2048);
    cvt(gen_w0, gw0, 168 * 256 * 16);
    cvt(gen_w1, gw1, 256 * 128 * 16);
    cvt(gen_w2, gw2, 128 * 3 * 16);

    // ---- encoder ----------------------------------------------------------
    {   // conv0: (B,3,128,128) -> (B,64,64,64)
        int M = BATCH * 64 * 64;
        conv_gemm_wmma<<<dim3(M / 64, 1), 128, 0, stream>>>(
            imgs16, cw0, cod_b0, raw, 3, 64, 128, ilog2i(64));
        bn_stats<<<64, 256, 0, stream>>>(raw, stats, 64, 64 * 64, ilog2i(64 * 64));
        int total = BATCH * 64 * 64 * 64;
        bn_act_f16<<<(total + 255) / 256, 256, 0, stream>>>(
            raw, stats, cod_g0, cod_be0, actB, 64, ilog2i(64 * 64), 1, total);
    }
    {   // conv1: (B,64,64,64) -> (B,128,32,32)
        int M = BATCH * 32 * 32;
        conv_gemm_wmma<<<dim3(M / 64, 2), 128, 0, stream>>>(
            actB, cw1, cod_b1, raw, 64, 128, 64, ilog2i(32));
        bn_stats<<<128, 256, 0, stream>>>(raw, stats, 128, 32 * 32, ilog2i(32 * 32));
        int total = BATCH * 128 * 32 * 32;
        bn_act_f16<<<(total + 255) / 256, 256, 0, stream>>>(
            raw, stats, cod_g1, cod_be1, actA, 128, ilog2i(32 * 32), 1, total);
    }
    {   // conv2: (B,128,32,32) -> (B,64,16,16), tanh -> Z_c (f16)
        int M = BATCH * 16 * 16;
        conv_gemm_wmma<<<dim3(M / 64, 1), 128, 0, stream>>>(
            actA, cw2, cod_b2, raw, 128, 64, 32, ilog2i(16));
        int total = BATCH * 64 * 256;
        tanh_to_f16<<<(total + 255) / 256, 256, 0, stream>>>(raw, Zc16, total);
    }

    // ---- wave params + latent assembly ------------------------------------
    mlp_wave_params<<<1, 64, 0, stream>>>(Zg, lW, lb, l1W, l1b, l2W, l2b, K1, K2);
    {
        int total = BATCH * 168 * 256;
        pack_latent<<<(total + 255) / 256, 256, 0, stream>>>(
            Zc16, Zl, Zg, K1, K2, phi, Z16, total);
    }

    // ---- decoder ----------------------------------------------------------
    {   // deconv0: (B,168,16,16) -> (B,256,32,32)
        int Mc = BATCH * 16 * 16;
        deconv_gemm_wmma<<<dim3(Mc / 64, 4, 4), 128, 0, stream>>>(
            Z16, gw0, gen_b0, raw, 168, 256, 16, ilog2i(16), 0);
        bn_stats<<<256, 256, 0, stream>>>(raw, stats, 256, 32 * 32, ilog2i(32 * 32));
        int total = BATCH * 256 * 32 * 32;
        bn_act_f16<<<(total + 255) / 256, 256, 0, stream>>>(
            raw, stats, gen_g0, gen_be0, actB, 256, ilog2i(32 * 32), 0, total);
    }
    {   // deconv1: (B,256,32,32) -> (B,128,64,64)
        int Mc = BATCH * 32 * 32;
        deconv_gemm_wmma<<<dim3(Mc / 64, 2, 4), 128, 0, stream>>>(
            actB, gw1, gen_b1, raw, 256, 128, 32, ilog2i(32), 0);
        bn_stats<<<128, 256, 0, stream>>>(raw, stats, 128, 64 * 64, ilog2i(64 * 64));
        int total = BATCH * 128 * 64 * 64;
        bn_act_f16<<<(total + 255) / 256, 256, 0, stream>>>(
            raw, stats, gen_g1, gen_be1, actA, 128, ilog2i(64 * 64), 0, total);
    }
    {   // deconv2: (B,128,64,64) -> (B,3,128,128), fused tanh -> d_out
        int Mc = BATCH * 64 * 64;
        deconv_gemm_wmma<<<dim3(Mc / 64, 1, 4), 128, 0, stream>>>(
            actA, gw2, gen_b2, (float*)d_out, 128, 3, 64, ilog2i(64), 1);
    }
}
